// WindowAttention_78855599554931
// MI455X (gfx1250) — matrix operand order repaired and checked by reading, not compiled, not run on hardware
//
#include <hip/hip_runtime.h>

#define HIDDEN   768
#define QKV_OUT  2304
#define NHEADS   12
#define HD       64
#define NWIN     1024
#define NTOK     64

typedef __attribute__((ext_vector_type(16))) _Float16 v16h;
typedef __attribute__((ext_vector_type(8)))  float    v8f;

union FragH { v16h v; _Float16 h[16]; unsigned u[8]; };

__device__ __forceinline__ int gtok_of(int win, int n) {
  int b = win >> 6, rem = win & 63;
  int wh = rem >> 3, ww = rem & 7;
  return b * 4096 + ((wh << 3) + (n >> 3)) * 64 + (ww << 3) + (n & 7);
}

// Convert float4 -> 4 x f16 and store 8 bytes to LDS.
__device__ __forceinline__ void store_h4(_Float16* dst, float4 v) {
  union { _Float16 h[4]; uint2 u; } p;
  p.h[0] = (_Float16)v.x; p.h[1] = (_Float16)v.y;
  p.h[2] = (_Float16)v.z; p.h[3] = (_Float16)v.w;
  *(uint2*)dst = p.u;
}

// A fragment (16x32 f16), row-major LDS tile with leading dim ldk.
__device__ __forceinline__ v16h load_frag_a(const _Float16* lds, int row0, int ldk) {
  int lane = threadIdx.x & 31;
  int m = lane & 15, half = lane >> 4;
  FragH f;
#pragma unroll
  for (int v = 0; v < 8; ++v) {
    int k = half * 16 + v * 2;
    f.u[v] = *(const unsigned*)(lds + (row0 + m) * ldk + k);
  }
  return f.v;
}

// B fragment (32x16 f16) from LDS stored [n][k] (output-channel-major).
__device__ __forceinline__ v16h load_frag_b(const _Float16* lds, int col0, int ldk) {
  int lane = threadIdx.x & 31;
  int n = lane & 15, half = lane >> 4;
  FragH f;
#pragma unroll
  for (int v = 0; v < 8; ++v) {
    int k = half * 16 + v * 2;
    f.u[v] = *(const unsigned*)(lds + (col0 + n) * ldk + k);
  }
  return f.v;
}

// B fragment from LDS stored [k][n] (k-major), leading dim ldn.
__device__ __forceinline__ v16h load_frag_b_km(const _Float16* lds, int col0, int ldn) {
  int lane = threadIdx.x & 31;
  int n = lane & 15, half = lane >> 4;
  FragH f;
#pragma unroll
  for (int v = 0; v < 8; ++v) {
    int kb = half * 16 + v * 2;
    f.h[2 * v]     = lds[kb * ldn + col0 + n];
    f.h[2 * v + 1] = lds[(kb + 1) * ldn + col0 + n];
  }
  return f.v;
}

// ---------------- Kernel 1: QKV GEMM + bias + RoPE --------------------------
// grid (18, 1024), block 128. Computes 64x128 tile of qkv for one window,
// stores q,k,v as f16 [s][win][head][n][d] (64x64 tiles), q pre-scaled 1/8.
__global__ __launch_bounds__(128)
void qkv_rope_kernel(const float* __restrict__ x,
                     const float* __restrict__ rope_cos,
                     const float* __restrict__ rope_sin,
                     const float* __restrict__ Wqkv,
                     const float* __restrict__ bqkv,
                     _Float16* __restrict__ qkv_ws) {
  __shared__ __align__(16) char smraw[64 * 128 * 4];     // 32 KB
  _Float16* sA    = (_Float16*)smraw;                    // 64x32 f16
  _Float16* sB    = (_Float16*)(smraw + 64 * 32 * 2);    // 128x32 f16
  float*    stage = (float*)smraw;                       // 64x128 f32 (reused)

  const int ntile = blockIdx.x;
  const int win   = blockIdx.y;
  const int t     = threadIdx.x;
  const int wave  = t >> 5;
  const int lane  = t & 31;
  const int nlane = lane & 15;
  const int half  = lane >> 4;

  v8f acc[4][2] = {};
  float4 ra[4], rb[8];   // prefetch registers (batched b128 global loads)

  auto load_a = [&](int k0) {
#pragma unroll
    for (int e = 0; e < 4; ++e) {           // 64x32 f32 = 512 float4
      int l = e * 128 + t;
      int row = l >> 3, kq = l & 7;
      ra[e] = *(const float4*)(x + (size_t)gtok_of(win, row) * HIDDEN + k0 + kq * 4);
    }
  };
  auto load_b = [&](int k0) {
#pragma unroll
    for (int e = 0; e < 8; ++e) {           // 128x32 f32 = 1024 float4
      int l = e * 128 + t;
      int n = l >> 3, kq = l & 7;
      rb[e] = *(const float4*)(Wqkv + (size_t)(ntile * 128 + n) * HIDDEN + k0 + kq * 4);
    }
  };

  load_a(0); load_b(0);
  for (int k0 = 0; k0 < HIDDEN; k0 += 32) {
    __syncthreads();
#pragma unroll
    for (int e = 0; e < 4; ++e) {
      int l = e * 128 + t; int row = l >> 3, kq = l & 7;
      store_h4(sA + row * 32 + kq * 4, ra[e]);
    }
#pragma unroll
    for (int e = 0; e < 8; ++e) {
      int l = e * 128 + t; int n = l >> 3, kq = l & 7;
      store_h4(sB + n * 32 + kq * 4, rb[e]);
    }
    __syncthreads();
    if (k0 + 32 < HIDDEN) { load_a(k0 + 32); load_b(k0 + 32); }  // overlap w/ WMMA

    v16h bfr[2];
#pragma unroll
    for (int ni = 0; ni < 2; ++ni) bfr[ni] = load_frag_b(sB, wave * 32 + ni * 16, 32);
#pragma unroll
    for (int mi = 0; mi < 4; ++mi) {
      v16h a = load_frag_a(sA, mi * 16, 32);
#pragma unroll
      for (int ni = 0; ni < 2; ++ni)
        acc[mi][ni] = __builtin_amdgcn_wmma_f32_16x16x32_f16(
            false, a, false, bfr[ni], (short)0, acc[mi][ni], false, false);
    }
  }

  __syncthreads();                                  // done reading sA/sB
#pragma unroll
  for (int mi = 0; mi < 4; ++mi)
#pragma unroll
    for (int ni = 0; ni < 2; ++ni)
#pragma unroll
      for (int r = 0; r < 8; ++r)
        stage[(mi * 16 + r + 8 * half) * 128 + wave * 32 + ni * 16 + nlane] = acc[mi][ni][r];
  __syncthreads();

  // RoPE + bias + store; thread t owns column t for all 64 rows.
  const int col = t;
  const int o = ntile * 128 + col;
  const int s = o / HIDDEN;                 // 0=q 1=k 2=v (tile never crosses s)
  const int h = (o % HIDDEN) / HD;
  const int d = o % HD;
  const float bo = bqkv[o];
  const int d2 = (d < 32) ? d + 32 : d - 32;
  const int col2 = col - d + d2;            // stays inside this 128-tile
  const float b2 = bqkv[ntile * 128 + col2];
  const float sgn = (d < 32) ? -1.f : 1.f;
  const float qscale = (s == 0) ? 0.125f : 1.f;   // 1/sqrt(64) folded into q
  _Float16* outbase = qkv_ws + ((size_t)(s * NWIN + win) * NHEADS + h) * 4096 + d;

  for (int r = 0; r < 64; ++r) {
    float val = stage[r * 128 + col] + bo;
    if (s < 2) {
      int gt = gtok_of(win, r);
      float c  = rope_cos[(size_t)gt * HD + d];
      float sn = rope_sin[(size_t)gt * HD + d];
      float v2 = stage[r * 128 + col2] + b2;
      val = val * c + sgn * v2 * sn;
    }
    outbase[r * 64] = (_Float16)(val * qscale);
  }
}

// ---------------- Kernel 2: attention per (window, head) --------------------
// grid (12, 1024), block 128 (4 waves, wave w owns rows 16w..16w+15).
// Q/K/V staged into LDS via the CDNA5 async copy engine (ASYNCcnt).
__global__ __launch_bounds__(128)
void attn_kernel(const _Float16* __restrict__ qkv_ws,
                 _Float16* __restrict__ attn_ws) {
  __shared__ __align__(16) _Float16 sm[4 * 4096];   // Q,K,V,P : 32 KB (LDS off 0)
  _Float16* sQ = sm;
  _Float16* sK = sm + 4096;
  _Float16* sV = sm + 8192;
  _Float16* sP = sm + 12288;

  const int head = blockIdx.x, win = blockIdx.y;
  const int t = threadIdx.x, wave = t >> 5, lane = t & 31;
  const int nlane = lane & 15, half = lane >> 4;

  const size_t base = ((size_t)win * NHEADS + head) * 4096;
  const size_t sstride = (size_t)NWIN * NHEADS * 4096;
  const uint4* gq = (const uint4*)(qkv_ws + base);
  const uint4* gk = (const uint4*)(qkv_ws + sstride + base);
  const uint4* gv = (const uint4*)(qkv_ws + 2 * sstride + base);

  // Async global->LDS DMA: 16 B per lane per op, no VGPR round-trip.
  // sm is the only __shared__ object => LDS byte offset of sm[i] is 2*i.
#pragma unroll
  for (int e = 0; e < 4; ++e) {
    unsigned dq = (unsigned)((t * 4 + e) * 16);   // byte offset of uint4 slot
    asm volatile("global_load_async_to_lds_b128 %0, %1, off"
                 :: "v"(dq),         "v"(gq + t * 4 + e) : "memory");
    asm volatile("global_load_async_to_lds_b128 %0, %1, off"
                 :: "v"(dq + 8192),  "v"(gk + t * 4 + e) : "memory");
    asm volatile("global_load_async_to_lds_b128 %0, %1, off"
                 :: "v"(dq + 16384), "v"(gv + t * 4 + e) : "memory");
  }
  asm volatile("s_wait_asynccnt 0x0" ::: "memory");
  __syncthreads();

  // S = Q K^T  (q pre-scaled by 1/8)
  v8f s[4] = {};
#pragma unroll
  for (int kb = 0; kb < 64; kb += 32) {
    v16h a = load_frag_a(sQ + kb, wave * 16, 64);
#pragma unroll
    for (int j = 0; j < 4; ++j) {
      v16h b = load_frag_b(sK + kb, j * 16, 64);   // implicit transpose of K
      s[j] = __builtin_amdgcn_wmma_f32_16x16x32_f16(
          false, a, false, b, (short)0, s[j], false, false);
    }
  }

  // softmax over the 64 columns: row = r + 8*half lives in 16-lane half-groups
#pragma unroll
  for (int r = 0; r < 8; ++r) {
    float mx = -3.402823466e38f;
#pragma unroll
    for (int j = 0; j < 4; ++j) mx = fmaxf(mx, s[j][r]);
#pragma unroll
    for (int off = 1; off < 16; off <<= 1) mx = fmaxf(mx, __shfl_xor(mx, off, 32));
    float sum = 0.f;
#pragma unroll
    for (int j = 0; j < 4; ++j) { float e = __expf(s[j][r] - mx); s[j][r] = e; sum += e; }
#pragma unroll
    for (int off = 1; off < 16; off <<= 1) sum += __shfl_xor(sum, off, 32);
    float inv = 1.f / (sum + 1e-9f);
#pragma unroll
    for (int j = 0; j < 4; ++j) s[j][r] *= inv;
  }

  // P -> per-wave LDS region (same-wave LDS ops are in-order; no barrier needed)
#pragma unroll
  for (int j = 0; j < 4; ++j)
#pragma unroll
    for (int r = 0; r < 8; ++r)
      sP[(wave * 16 + r + 8 * half) * 64 + j * 16 + nlane] = (_Float16)s[j][r];

  // O = P @ V
  v8f o[4] = {};
#pragma unroll
  for (int kb = 0; kb < 64; kb += 32) {
    v16h a = load_frag_a(sP + kb, wave * 16, 64);
#pragma unroll
    for (int j = 0; j < 4; ++j) {
      v16h b = load_frag_b_km(sV + kb * 64, j * 16, 64);
      o[j] = __builtin_amdgcn_wmma_f32_16x16x32_f16(
          false, a, false, b, (short)0, o[j], false, false);
    }
  }

  // attn_ws[win][row][head*64 + d] (f16)
#pragma unroll
  for (int j = 0; j < 4; ++j)
#pragma unroll
    for (int r = 0; r < 8; ++r) {
      int row = wave * 16 + r + 8 * half;
      attn_ws[(size_t)win * NTOK * HIDDEN + (size_t)row * HIDDEN + head * 64 + j * 16 + nlane]
          = (_Float16)o[j][r];
    }
}

// ---------------- Kernel 3: proj GEMM + bias + un-partition -----------------
// grid (6, 1024), block 128. out rows scattered back to original token order.
__global__ __launch_bounds__(128)
void proj_kernel(const _Float16* __restrict__ attn_ws,
                 const float* __restrict__ Wproj,
                 const float* __restrict__ bproj,
                 float* __restrict__ out) {
  __shared__ __align__(16) char smraw[(64 * 32 + 128 * 32) * 2];  // 12 KB
  _Float16* sA = (_Float16*)smraw;
  _Float16* sB = (_Float16*)(smraw + 64 * 32 * 2);

  const int otile = blockIdx.x, win = blockIdx.y;
  const int t = threadIdx.x, wave = t >> 5, lane = t & 31;
  const int nlane = lane & 15, half = lane >> 4;

  const uint4* gA = (const uint4*)(attn_ws + (size_t)win * NTOK * HIDDEN);
  v8f acc[4][2] = {};
  uint4  ra[2];
  float4 rb[8];

  auto load_a = [&](int k0) {
#pragma unroll
    for (int e = 0; e < 2; ++e) {           // 64x32 f16 = 256 uint4
      int l = e * 128 + t;
      int row = l >> 2, q = l & 3;          // 4 uint4 per 32-half row slice
      ra[e] = gA[(size_t)row * (HIDDEN / 8) + (k0 >> 3) + q];
    }
  };
  auto load_b = [&](int k0) {
#pragma unroll
    for (int e = 0; e < 8; ++e) {           // 128x32 f32 = 1024 float4
      int l = e * 128 + t;
      int n = l >> 3, kq = l & 7;
      rb[e] = *(const float4*)(Wproj + (size_t)(otile * 128 + n) * HIDDEN + k0 + kq * 4);
    }
  };

  load_a(0); load_b(0);
  for (int k0 = 0; k0 < HIDDEN; k0 += 32) {
    __syncthreads();
    uint4* sA4 = (uint4*)sA;
#pragma unroll
    for (int e = 0; e < 2; ++e) {
      int l = e * 128 + t; int row = l >> 2, q = l & 3;
      sA4[row * 4 + q] = ra[e];
    }
#pragma unroll
    for (int e = 0; e < 8; ++e) {
      int l = e * 128 + t; int n = l >> 3, kq = l & 7;
      store_h4(sB + n * 32 + kq * 4, rb[e]);
    }
    __syncthreads();
    if (k0 + 32 < HIDDEN) { load_a(k0 + 32); load_b(k0 + 32); }  // overlap w/ WMMA

    v16h bfr[2];
#pragma unroll
    for (int ni = 0; ni < 2; ++ni) bfr[ni] = load_frag_b(sB, wave * 32 + ni * 16, 32);
#pragma unroll
    for (int mi = 0; mi < 4; ++mi) {
      v16h a = load_frag_a(sA, mi * 16, 32);
#pragma unroll
      for (int ni = 0; ni < 2; ++ni)
        acc[mi][ni] = __builtin_amdgcn_wmma_f32_16x16x32_f16(
            false, a, false, bfr[ni], (short)0, acc[mi][ni], false, false);
    }
  }

#pragma unroll
  for (int mi = 0; mi < 4; ++mi)
#pragma unroll
    for (int ni = 0; ni < 2; ++ni) {
      int o = otile * 128 + wave * 32 + ni * 16 + nlane;
      float bo = bproj[o];
#pragma unroll
      for (int r = 0; r < 8; ++r) {
        int row = mi * 16 + r + 8 * half;
        int gt = gtok_of(win, row);
        out[(size_t)gt * HIDDEN + o] = acc[mi][ni][r] + bo;
      }
    }
}

extern "C" void kernel_launch(void* const* d_in, const int* in_sizes, int n_in,
                              void* d_out, int out_size, void* d_ws, size_t ws_size,
                              hipStream_t stream) {
  const float* x        = (const float*)d_in[0];
  const float* rope_cos = (const float*)d_in[1];
  const float* rope_sin = (const float*)d_in[2];
  const float* Wqkv     = (const float*)d_in[3];
  const float* bqkv     = (const float*)d_in[4];
  const float* Wproj    = (const float*)d_in[5];
  const float* bproj    = (const float*)d_in[6];

  _Float16* qkv_ws  = (_Float16*)d_ws;                               // 302 MB
  _Float16* attn_ws = qkv_ws + (size_t)3 * NWIN * NHEADS * 4096;     // 100 MB

  qkv_rope_kernel<<<dim3(QKV_OUT / 128, NWIN), 128, 0, stream>>>(
      x, rope_cos, rope_sin, Wqkv, bqkv, qkv_ws);
  attn_kernel<<<dim3(NHEADS, NWIN), 128, 0, stream>>>(qkv_ws, attn_ws);
  proj_kernel<<<dim3(HIDDEN / 128, NWIN), 128, 0, stream>>>(
      attn_ws, Wproj, bproj, (float*)d_out);
}
